// Mex_66400194396566
// MI455X (gfx1250) — matrix operand order repaired and checked by reading, not compiled, not run on hardware
//
#include <hip/hip_runtime.h>

// ---------------------------------------------------------------------------
// MEX pooling: out[b,m,h,w] = T + m_b[m] + log( sum_k exp(x_k - T)*Eb[m,k] ) - ln(576)
// B=16, C=64, H=W=128, BH=BW=3, M=16, n=576, Ho=Wo=126, EPS=1
// K-order chosen as k = (dh*3+dw)*64 + c so every 32-wide K chunk is one
// (dh,dw) tap over 32 consecutive channels -> A/B fragments are contiguous
// b128 LDS loads with compile-time offsets. GEMM: v_wmma_f32_16x16x32_bf16.
// ---------------------------------------------------------------------------

typedef __attribute__((ext_vector_type(16))) __bf16 v16bf;
typedef __attribute__((ext_vector_type(8)))  float  v8f;

#define CPAD 72            // padded channel stride of exp-region rows (bank spread)
#define EBP  584           // padded K stride of Eb rows (bank spread)
#define WAVES 4

__device__ __forceinline__ unsigned bfbits(float f) {
    unsigned u = __builtin_bit_cast(unsigned, f);
    return (u + 0x7FFFu + ((u >> 16) & 1u)) >> 16;     // RNE to bf16 bits
}
__device__ __forceinline__ unsigned pk2(float lo, float hi) {
    return (bfbits(lo) & 0xFFFFu) | (bfbits(hi) << 16);
}

// ---------------------------------------------------------------------------
// Pass A: channel max map  cmax[b,h,w] = max_c x[b,c,h,w]   (coalesced sweep)
// ---------------------------------------------------------------------------
__global__ __launch_bounds__(256) void mex_cmax(const float* __restrict__ x,
                                                float* __restrict__ cmax) {
    const int i  = blockIdx.x * 256 + threadIdx.x;     // 0 .. 16*128*128-1
    const int b  = i >> 14;
    const int hw = i & 16383;
    const float* p = x + (size_t)b * (64 * 16384) + hw;
    float m = p[0];
    #pragma unroll
    for (int c = 1; c < 64; ++c) m = fmaxf(m, p[(size_t)c * 16384]);
    cmax[i] = m;
}

// ---------------------------------------------------------------------------
// Pass B: m_b[m] = max_k b[m,k];  Eb[m, k'=(t*64+c)] = bf16(exp(b - m_b)),
// stored with padded row stride EBP in the reordered K layout.
// ---------------------------------------------------------------------------
__global__ __launch_bounds__(64) void mex_prep(const float* __restrict__ off,
                                               __bf16* __restrict__ Eb,
                                               float* __restrict__ mb) {
    const int m = blockIdx.x;           // 0..15
    const int t = threadIdx.x;          // 0..63
    __shared__ float red[64];
    float mx = -3.0e38f;
    for (int k = t; k < 576; k += 64) mx = fmaxf(mx, off[m * 576 + k]);
    red[t] = mx;
    __syncthreads();
    for (int s = 32; s > 0; s >>= 1) {
        if (t < s) red[t] = fmaxf(red[t], red[t + s]);
        __syncthreads();
    }
    const float M = red[0];
    if (t == 0) mb[m] = M;
    for (int s = t; s < 576; s += 64) {
        const int c  = s & 63;          // storage order: k' = tap*64 + c
        const int tp = s >> 6;          // tap = dh*3+dw
        const float v = off[m * 576 + c * 9 + tp];
        Eb[m * EBP + s] = __builtin_bit_cast(__bf16, (unsigned short)bfbits(__expf(v - M)));
    }
}

// ---------------------------------------------------------------------------
// Main: 4 waves/block, one 16-wide output tile per wave.
// Tiles: b in 0..15, h0 in 0..125, tw in 0..7 (w0=16*tw) -> 16128 tiles.
// ---------------------------------------------------------------------------
__global__ __launch_bounds__(128) void mex_main(const float* __restrict__ x,
                                                const float* __restrict__ cmax,
                                                const __bf16* __restrict__ EbG,
                                                const float* __restrict__ mbG,
                                                float* __restrict__ out) {
    __shared__ __bf16 EbS[16 * EBP];           // 18.25 KB
    __shared__ float  mbS[16];
    __shared__ __bf16 eS[WAVES][3 * 18 * CPAD]; // 4 * 7.6 KB, layout [dh][w][c]

    const int tid = threadIdx.x;

    // Block-wide preload of Eb (16*EBP bf16 = 1168 uint4) and m_b.
    {
        const uint4* src = (const uint4*)EbG;
        uint4*       dst = (uint4*)EbS;
        for (int i = tid; i < (16 * EBP * 2) / 16; i += 128) dst[i] = src[i];
        if (tid < 16) mbS[tid] = mbG[tid];
    }
    __syncthreads();

    const int wave = tid >> 5;
    const int lane = tid & 31;
    const int half = lane >> 4;    // ISA fragment lane-group
    const int row  = lane & 15;

    const int tile = blockIdx.x * WAVES + wave;     // 0..16127 exactly
    const int b    = tile / (126 * 8);
    const int rem  = tile % (126 * 8);
    const int h0   = rem >> 3;
    const int w0   = (rem & 7) << 4;

    const float* xb = x + (size_t)b * (64 * 16384);
    __bf16* eW = &eS[wave][0];

    // --- tile stabilizer from cmax map: max over 3x18 window -----------------
    float tmax = -3.0e38f;
    #pragma unroll
    for (int it = 0; it < 2; ++it) {
        const int j = it * 32 + lane;
        if (j < 54) {
            const int dh = j / 18;
            const int w  = j - dh * 18;
            int wc = w0 + w; wc = wc < 127 ? wc : 127;
            tmax = fmaxf(tmax, cmax[b * 16384 + (h0 + dh) * 128 + wc]);
        }
    }
    #pragma unroll
    for (int o = 16; o > 0; o >>= 1) tmax = fmaxf(tmax, __shfl_xor(tmax, o, 32));

    // --- fill exp-region: lane owns one (dh,w) pair, 64 channels, b128 stores
    #pragma unroll
    for (int it = 0; it < 2; ++it) {
        const int pair = it * 32 + lane;
        if (pair < 54) {
            const int dh = pair / 18;
            const int w  = pair - dh * 18;
            int wc = w0 + w; wc = wc < 127 ? wc : 127;       // clamp halo
            const float* g = xb + (h0 + dh) * 128 + wc;      // + c*16384
            uint4* dst = (uint4*)(eW + pair * CPAD);
            #pragma unroll
            for (int cb = 0; cb < 8; ++cb) {
                float f[8];
                #pragma unroll
                for (int u = 0; u < 8; ++u)
                    f[u] = __expf(g[(size_t)(cb * 8 + u) * 16384] - tmax);
                uint4 q;
                q.x = pk2(f[0], f[1]); q.y = pk2(f[2], f[3]);
                q.z = pk2(f[4], f[5]); q.w = pk2(f[6], f[7]);
                dst[cb] = q;
            }
        }
    }
    // Cross-lane LDS handoff within the wave.
    asm volatile("s_wait_dscnt 0x0" ::: "memory");
    __builtin_amdgcn_wave_barrier();

    // --- GEMM: 9 taps x 2 channel-halves, all offsets compile-time ----------
    v8f acc = {};
    const __bf16* aBase = eW  + row * CPAD + half * 8;    // A: row=pixel
    const __bf16* bBase = EbS + row * EBP  + half * 8;    // B: col=m instance
    #pragma unroll
    for (int t = 0; t < 9; ++t) {
        const int dh = t / 3, dw = t % 3;                 // constants after unroll
        const int arow = (dh * 18 + dw) * CPAD;
        #pragma unroll
        for (int cb2 = 0; cb2 < 2; ++cb2) {
            const int cbase = cb2 * 32;
            union { v16bf v; uint4 q[2]; } au, bu;
            au.q[0] = *(const uint4*)(aBase + arow + cbase);
            au.q[1] = *(const uint4*)(aBase + arow + cbase + 16);
            bu.q[0] = *(const uint4*)(bBase + t * 64 + cbase);
            bu.q[1] = *(const uint4*)(bBase + t * 64 + cbase + 16);
            acc = __builtin_amdgcn_wmma_f32_16x16x32_bf16(
                false, au.v, false, bu.v, (short)0, acc, false, false);
        }
    }

    // --- epilogue: D VGPR r -> (pixel = w0 + r + 8*half, m = lane&15) -------
    const float base = tmax + mbS[row] - 6.3561076606830749f;   // - ln(576)
    const size_t obase = (((size_t)b * 16 + row) * 126 + h0) * 126;
    #pragma unroll
    for (int r = 0; r < 8; ++r) {
        const int w = w0 + r + half * 8;
        if (w < 126) out[obase + w] = base + __logf(acc[r]);
    }
}

// ---------------------------------------------------------------------------
extern "C" void kernel_launch(void* const* d_in, const int* in_sizes, int n_in,
                              void* d_out, int out_size, void* d_ws, size_t ws_size,
                              hipStream_t stream) {
    const float* x   = (const float*)d_in[0];   // (16,64,128,128) f32
    const float* off = (const float*)d_in[1];   // (1,16,64,3,3)  f32

    // workspace layout
    float*  cmax = (float*)d_ws;                                  // 1 MB
    __bf16* Eb   = (__bf16*)((char*)d_ws + 16 * 16384 * 4);       // 16*EBP bf16
    float*  mb   = (float*)((char*)Eb + 16 * EBP * 2);            // 16 f32

    mex_cmax<<<1024, 256, 0, stream>>>(x, cmax);
    mex_prep<<<16, 64, 0, stream>>>(off, Eb, mb);
    mex_main<<<4032, 128, 0, stream>>>(x, cmax, Eb, mb, (float*)d_out);
}
